// NTMReadHead_90400471646262
// MI455X (gfx1250) — compile-verified, matrix operand order checked by reading, not run on hardware
//
#include <hip/hip_runtime.h>
#include <hip/hip_bf16.h>
#include <math.h>

#define BB   256
#define NN   8192
#define MM   128
#define CC   1024
#define PCOL 144           // 128 (k) + 1 (beta) + 1 (g) + 3 (s) + 1 (gamma) + 10 pad
#define EPSV 1e-16f

typedef __attribute__((ext_vector_type(16))) _Float16 v16h;
typedef __attribute__((ext_vector_type(8)))  _Float16 v8h;
typedef __attribute__((ext_vector_type(8)))  float    v8f;
typedef __attribute__((ext_vector_type(4)))  float    v4f;

__device__ __forceinline__ float softplusf(float x) {
    return (x > 20.f) ? x : log1pf(expf(x));
}

__device__ __forceinline__ float waveReduceSum(float v) {
#pragma unroll
    for (int m = 16; m > 0; m >>= 1) v += __shfl_xor(v, m, 32);
    return v;
}

// Non-temporal 128-bit load: memory tensor has zero reuse (1 GiB stream,
// L2 = 192 MB), so keep it out of the cache-resident set.  Uses a clang
// ext-vector type, which __builtin_nontemporal_load accepts.
__device__ __forceinline__ v4f nt_load4(const float* p) {
    return __builtin_nontemporal_load((const v4f*)p);
}

// ---------------------------------------------------------------------------
// Kernel 1: convert embeddings to f16; stack+convert the 5 head weight
// matrices into one padded [PCOL x CC] f16 matrix.
// ---------------------------------------------------------------------------
__global__ void pack_kernel(const float* __restrict__ emb,
                            const float* __restrict__ Wk,
                            const float* __restrict__ Wb,
                            const float* __restrict__ Wg,
                            const float* __restrict__ Ws,
                            const float* __restrict__ Wgam,
                            _Float16* __restrict__ Ef,
                            _Float16* __restrict__ Wf) {
    int i = blockIdx.x * 256 + threadIdx.x;
    const int NE = BB * CC;
    if (i < NE) { Ef[i] = (_Float16)emb[i]; return; }
    int j = i - NE;
    if (j >= PCOL * CC) return;
    int r = j >> 10, c = j & (CC - 1);
    float v = 0.f;
    if      (r < 128)  v = Wk[r * CC + c];
    else if (r == 128) v = Wb[c];
    else if (r == 129) v = Wg[c];
    else if (r < 133)  v = Ws[(r - 130) * CC + c];
    else if (r == 133) v = Wgam[c];
    Wf[j] = (_Float16)v;
}

// ---------------------------------------------------------------------------
// Kernel 2: projection GEMM  proj[256 x 144] = E[256x1024] * Wall^T[1024x144]
// via v_wmma_f32_16x16x32_f16.  One wave per 16x16 D tile, K-loop of 32.
// A layout (16x32 f16): lane L holds row M=L%16; halves 0..7 = K base+hi8..+7,
// halves 8..15 = K base+16+hi8..+7, hi8 = 8*(L>=16).  B mirrored (lane = col).
// ---------------------------------------------------------------------------
__global__ __launch_bounds__(32) void proj_gemm_kernel(
        const _Float16* __restrict__ E,
        const _Float16* __restrict__ W,
        float* __restrict__ proj) {
    const int lane = threadIdx.x;
    const int n16  = lane & 15;
    const int hi8  = (lane >> 4) << 3;
    const _Float16* Arow = E + (size_t)(blockIdx.x * 16 + n16) * CC;
    const _Float16* Brow = W + (size_t)(blockIdx.y * 16 + n16) * CC;
    v8f acc = {};
    for (int kc = 0; kc < CC; kc += 32) {
        v16h a, b;
        ((v8h*)&a)[0] = *(const v8h*)(Arow + kc + hi8);
        ((v8h*)&a)[1] = *(const v8h*)(Arow + kc + 16 + hi8);
        ((v8h*)&b)[0] = *(const v8h*)(Brow + kc + hi8);
        ((v8h*)&b)[1] = *(const v8h*)(Brow + kc + 16 + hi8);
        acc = __builtin_amdgcn_wmma_f32_16x16x32_f16(
            false, a, false, b, (short)0, acc, false, false);
    }
    // D: lane holds col N = lane%16; VGPR r holds row M = r + 8*(lane>=16)
    const int mbase = (lane >> 4) << 3;
    float* out = proj + (size_t)(blockIdx.x * 16) * PCOL + blockIdx.y * 16 + n16;
#pragma unroll
    for (int r = 0; r < 8; ++r)
        out[(size_t)(mbase + r) * PCOL] = acc[r];
}

// ---------------------------------------------------------------------------
// Kernel 3: per-batch head epilogue: k_e = k + bk + EPS, ||k_e||, and the four
// scalar heads with their activations.  One 128-thread block per batch.
// ---------------------------------------------------------------------------
__global__ __launch_bounds__(128) void heads_kernel(
        const float* __restrict__ proj,
        const float* __restrict__ bk, const float* __restrict__ bb,
        const float* __restrict__ bg, const float* __restrict__ bs,
        const float* __restrict__ bgam,
        float* __restrict__ ke, float* __restrict__ beta,
        float* __restrict__ gate, float* __restrict__ shead,
        float* __restrict__ gamma, float* __restrict__ normk) {
    int b = blockIdx.x, t = threadIdx.x;
    __shared__ float red[128];
    float v = proj[(size_t)b * PCOL + t] + bk[t] + EPSV;
    ke[(size_t)b * MM + t] = v;
    red[t] = v * v;
    __syncthreads();
    for (int s = 64; s > 0; s >>= 1) {
        if (t < s) red[t] += red[t + s];
        __syncthreads();
    }
    if (t == 0) {
        const float* pb = proj + (size_t)b * PCOL;
        normk[b] = sqrtf(red[0]);
        beta[b]  = softplusf(pb[128] + bb[0]);
        gate[b]  = 1.f / (1.f + expf(-(pb[129] + bg[0])));
        float s0 = pb[130] + bs[0], s1 = pb[131] + bs[1], s2 = pb[132] + bs[2];
        float mx = fmaxf(s0, fmaxf(s1, s2));
        float e0 = expf(s0 - mx), e1 = expf(s1 - mx), e2 = expf(s2 - mx);
        float ss = e0 + e1 + e2;
        shead[b * 3 + 0] = e0 / ss;
        shead[b * 3 + 1] = e1 / ss;
        shead[b * 3 + 2] = e2 / ss;
        gamma[b] = 1.f + softplusf(pb[133] + bgam[0]);
    }
}

// ---------------------------------------------------------------------------
// Kernel 4: streaming pass 1 over memory (1 GiB, non-temporal): fused dot +
// row-norm + cosine sim + beta scaling.  One wave per 512B row; 8 waves/block,
// 256 rows/block, grid (N/256, B).
// ---------------------------------------------------------------------------
__global__ __launch_bounds__(256) void sim_kernel(
        const float* __restrict__ mem, const float* __restrict__ ke,
        const float* __restrict__ beta, const float* __restrict__ normk,
        float* __restrict__ z) {
    int b = blockIdx.y;
    int tid = threadIdx.x, lane = tid & 31, wid = tid >> 5;
    v4f kv = *(const v4f*)(ke + (size_t)b * MM + lane * 4);
    float nk = normk[b], bt = beta[b];
    int base = blockIdx.x * 256 + wid * 32;
    const float* mb = mem + ((size_t)b * NN + base) * MM;
    for (int i = 0; i < 32; ++i) {
        const float* mr = mb + (size_t)i * MM;
        if (i + 1 < 32)
            __builtin_prefetch(mr + MM + lane * 4, 0, 0);   // global_prefetch_b8
        v4f mv = nt_load4(mr + lane * 4);
        float mx = mv.x + EPSV, my = mv.y + EPSV,
              mz = mv.z + EPSV, mw = mv.w + EPSV;
        float d = mx * kv.x + my * kv.y + mz * kv.z + mw * kv.w;
        float q = mx * mx + my * my + mz * mz + mw * mw;
        d = waveReduceSum(d);
        q = waveReduceSum(q);
        if (lane == 0) {
            float denom = fmaxf(sqrtf(q) * nk, 1e-8f);
            z[(size_t)b * NN + base + i] = bt * (d / denom);
        }
    }
}

// ---------------------------------------------------------------------------
// Kernel 5: one block per batch: softmax over N=8192, gate interpolation,
// 3-tap circular shift (LDS-resident row, 32KB), gamma sharpening, renorm.
// ---------------------------------------------------------------------------
__global__ __launch_bounds__(256) void weight_kernel(
        const float* __restrict__ z, const float* __restrict__ wprev,
        const float* __restrict__ gate, const float* __restrict__ shead,
        const float* __restrict__ gamma, float* __restrict__ wout) {
    int b = blockIdx.x, t = threadIdx.x;
    __shared__ float red[256];
    __shared__ float wgs[NN];
    const float* zb = z + (size_t)b * NN;
    float zl[32];
    float lmax = -3.0e38f;
#pragma unroll
    for (int i = 0; i < 32; ++i) {
        zl[i] = zb[t + i * 256];
        lmax = fmaxf(lmax, zl[i]);
    }
    red[t] = lmax; __syncthreads();
    for (int s = 128; s > 0; s >>= 1) {
        if (t < s) red[t] = fmaxf(red[t], red[t + s]);
        __syncthreads();
    }
    float zmax = red[0]; __syncthreads();
    float lsum = 0.f;
#pragma unroll
    for (int i = 0; i < 32; ++i) { zl[i] = expf(zl[i] - zmax); lsum += zl[i]; }
    red[t] = lsum; __syncthreads();
    for (int s = 128; s > 0; s >>= 1) {
        if (t < s) red[t] += red[t + s];
        __syncthreads();
    }
    float S1 = red[0]; __syncthreads();
    float gb = gate[b];
    const float* wp = wprev + (size_t)b * NN;
#pragma unroll
    for (int i = 0; i < 32; ++i) {
        int n = t + i * 256;
        wgs[n] = gb * (zl[i] / S1) + (1.f - gb) * wp[n];
    }
    __syncthreads();
    float s0 = shead[b * 3], s1 = shead[b * 3 + 1], s2 = shead[b * 3 + 2];
    float gam = gamma[b];
    float sharp[32];
    float lsum2 = 0.f;
#pragma unroll
    for (int i = 0; i < 32; ++i) {
        int n  = t + i * 256;
        int nm = (n == 0) ? (NN - 1) : (n - 1);
        int np = (n == NN - 1) ? 0 : (n + 1);
        float wsft = s0 * wgs[nm] + s1 * wgs[n] + s2 * wgs[np];
        sharp[i] = powf(wsft, gam);
        lsum2 += sharp[i];
    }
    red[t] = lsum2; __syncthreads();
    for (int s = 128; s > 0; s >>= 1) {
        if (t < s) red[t] += red[t + s];
        __syncthreads();
    }
    float S2 = red[0] + EPSV;
    float* wo = wout + (size_t)b * NN;
#pragma unroll
    for (int i = 0; i < 32; ++i)
        wo[t + i * 256] = sharp[i] / S2;
}

// ---------------------------------------------------------------------------
// Kernel 6: zero the r output region (read kernel accumulates with atomics).
// ---------------------------------------------------------------------------
__global__ void zero_kernel(float* __restrict__ p, int n) {
    int i = blockIdx.x * 256 + threadIdx.x;
    if (i < n) p[i] = 0.f;
}

// ---------------------------------------------------------------------------
// Kernel 7: streaming pass 2 over memory (1 GiB, non-temporal):
// r[b,:] = sum_n w[b,n]*mem.  Same wave-per-row layout; LDS partial
// (ds_add_f32) + one global f32 atomic per (block, m).  w (8 MB) stays
// regular-temporal: it is the L2-resident reuse set.
// ---------------------------------------------------------------------------
__global__ __launch_bounds__(256) void read_kernel(
        const float* __restrict__ mem, const float* __restrict__ w,
        float* __restrict__ r) {
    int b = blockIdx.y;
    int tid = threadIdx.x, lane = tid & 31, wid = tid >> 5;
    __shared__ float rp[MM];
    if (tid < MM) rp[tid] = 0.f;
    __syncthreads();
    int base = blockIdx.x * 256 + wid * 32;
    const float* mb = mem + ((size_t)b * NN + base) * MM;
    const float* wb = w + (size_t)b * NN + base;
    float a0 = 0.f, a1 = 0.f, a2 = 0.f, a3 = 0.f;
    for (int i = 0; i < 32; ++i) {
        const float* mr = mb + (size_t)i * MM;
        if (i + 1 < 32)
            __builtin_prefetch(mr + MM + lane * 4, 0, 0);
        float wv = wb[i];
        v4f mv = nt_load4(mr + lane * 4);
        a0 += wv * mv.x; a1 += wv * mv.y; a2 += wv * mv.z; a3 += wv * mv.w;
    }
    atomicAdd(&rp[lane * 4 + 0], a0);
    atomicAdd(&rp[lane * 4 + 1], a1);
    atomicAdd(&rp[lane * 4 + 2], a2);
    atomicAdd(&rp[lane * 4 + 3], a3);
    __syncthreads();
    if (tid < MM) atomicAdd(&r[(size_t)b * MM + tid], rp[tid]);
}

// ---------------------------------------------------------------------------
extern "C" void kernel_launch(void* const* d_in, const int* in_sizes, int n_in,
                              void* d_out, int out_size, void* d_ws, size_t ws_size,
                              hipStream_t stream) {
    const float* emb   = (const float*)d_in[0];
    const float* wprev = (const float*)d_in[1];
    const float* mem   = (const float*)d_in[2];
    const float* Wk    = (const float*)d_in[3];
    const float* bk    = (const float*)d_in[4];
    const float* Wb    = (const float*)d_in[5];
    const float* bb    = (const float*)d_in[6];
    const float* Wg    = (const float*)d_in[7];
    const float* bg    = (const float*)d_in[8];
    const float* Ws    = (const float*)d_in[9];
    const float* bs    = (const float*)d_in[10];
    const float* Wgam  = (const float*)d_in[11];
    const float* bgam  = (const float*)d_in[12];

    float* out_r = (float*)d_out;                 // [B, M]
    float* out_w = out_r + (size_t)BB * MM;       // [B, N]

    char* p = (char*)d_ws;
    float*    z     = (float*)p;     p += (size_t)BB * NN * 4;   // 8 MiB
    _Float16* Ef    = (_Float16*)p;  p += (size_t)BB * CC * 2;
    _Float16* Wf    = (_Float16*)p;  p += (size_t)PCOL * CC * 2;
    float*    proj  = (float*)p;     p += (size_t)BB * PCOL * 4;
    float*    ke    = (float*)p;     p += (size_t)BB * MM * 4;
    float*    beta  = (float*)p;     p += (size_t)BB * 4;
    float*    gate  = (float*)p;     p += (size_t)BB * 4;
    float*    gamma = (float*)p;     p += (size_t)BB * 4;
    float*    normk = (float*)p;     p += (size_t)BB * 4;
    float*    shead = (float*)p;     p += (size_t)BB * 3 * 4;

    // 1) f32 -> f16 pack of E and stacked weights
    {
        int total = BB * CC + PCOL * CC;
        pack_kernel<<<(total + 255) / 256, 256, 0, stream>>>(
            emb, Wk, Wb, Wg, Ws, Wgam, Ef, Wf);
    }
    // 2) WMMA projection GEMM: proj[256 x 144]
    proj_gemm_kernel<<<dim3(BB / 16, PCOL / 16), 32, 0, stream>>>(Ef, Wf, proj);
    // 3) head activations + ||k_e||
    heads_kernel<<<BB, 128, 0, stream>>>(proj, bk, bb, bg, bs, bgam,
                                         ke, beta, gate, shead, gamma, normk);
    // 4) pass 1 over memory: z = beta * cosine(mem+eps, k+eps)
    sim_kernel<<<dim3(NN / 256, BB), 256, 0, stream>>>(mem, ke, beta, normk, z);
    // 5) softmax / gate / shift / sharpen -> w  (written into d_out)
    weight_kernel<<<BB, 256, 0, stream>>>(z, wprev, gate, shead, gamma, out_w);
    // 6) zero r, then pass 2 over memory: r = w . mem
    zero_kernel<<<(BB * MM + 255) / 256, 256, 0, stream>>>(out_r, BB * MM);
    read_kernel<<<dim3(NN / 256, BB), 256, 0, stream>>>(mem, out_w, out_r);
}
